// RNNDecoder_31756988186968
// MI455X (gfx1250) — matrix operand order repaired and checked by reading, not compiled, not run on hardware
//
#include <hip/hip_runtime.h>
#include <hip/hip_bf16.h>
#include <math.h>

// ---------------------------------------------------------------------------
// Persistent per-wave GRU decoder for MI455X (gfx1250, wave32, WMMA f16).
// Each wave owns 16 batch rows and runs the full T=500 recurrence locally.
// All GEMMs use v_wmma_f32_16x16x32_f16 with fp16 weights staged in LDS.
// Activations use raw v_exp_f32/v_rcp_f32 (branch-free) instead of ocml.
// ---------------------------------------------------------------------------

typedef _Float16 half8 __attribute__((ext_vector_type(8)));
typedef _Float16 v16h  __attribute__((ext_vector_type(16)));
typedef float    v8f   __attribute__((ext_vector_type(8)));

namespace {
constexpr int BATCH  = 2048;
constexpr int TSTEPS = 500;
constexpr int LDIM   = 128;          // latent dim
constexpr int DDIM   = 64;           // output dim
constexpr int UDIM   = 128;          // mlp hidden
constexpr int GDIM   = 3 * LDIM;     // 384 gate rows
constexpr int WAVES  = 4;            // waves per workgroup
constexpr int WROWS  = 16;           // batch rows per wave
constexpr int WGROWS = WAVES * WROWS; // 64 rows per workgroup
}

__device__ __forceinline__ v8f wmma16(v16h a, v16h b, v8f c) {
  // (neg_a, A, neg_b, B, c_mod, C, reuse_a, reuse_b)
  return __builtin_amdgcn_wmma_f32_16x16x32_f16(false, a, false, b, (short)0, c,
                                                false, false);
}

// A operand: 16x32 f16 tile from row-major [16][stride] fp16 LDS buffer.
// lanes 0-15: M=lane, K = kbase+0..7 | kbase+16..23
// lanes 16-31: M=lane-16, K = kbase+8..15 | kbase+24..31
__device__ __forceinline__ v16h load_A(const _Float16* p, int stride, int kbase,
                                       int lane) {
  const int m  = lane & 15;
  const int kA = kbase + ((lane & 16) ? 8 : 0);
  const half8 lo = *(const half8*)(p + m * stride + kA);
  const half8 hi = *(const half8*)(p + m * stride + kA + 16);
  return __builtin_shufflevector(lo, hi, 0, 1, 2, 3, 4, 5, 6, 7,
                                 8, 9, 10, 11, 12, 13, 14, 15);
}

// B operand: 32x16 f16 tile; B[k][n] = W[nbase+n][kbase+k] for row-major W[n][k].
// lanes 0-15: N=lane, K=kbase+0..7 | kbase+16..23; lanes 16-31: N=lane-16, K=kbase+8..15 | kbase+24..31.
__device__ __forceinline__ v16h load_B(const _Float16* p, int stride, int nbase,
                                       int kbase, int lane) {
  const int n = nbase + (lane & 15);
  const int k = kbase + ((lane & 16) ? 8 : 0);
  const _Float16* q = p + n * stride + k;
  const half8 lo = *(const half8*)(q);
  const half8 hi = *(const half8*)(q + 16);
  return __builtin_shufflevector(lo, hi, 0, 1, 2, 3, 4, 5, 6, 7,
                                 8, 9, 10, 11, 12, 13, 14, 15);
}

// Branch-free sigmoid on raw HW transcendentals:
//   sigmoid(x) = rcp(1 + exp2(-x*log2(e)))
// exp2 saturates to 0/+inf and rcp(+inf)=0, so no range fixup is needed.
__device__ __forceinline__ float fast_sigmoid(float x) {
  return __builtin_amdgcn_rcpf(
      1.0f + __builtin_amdgcn_exp2f(-1.442695040888963f * x));
}

// tanh(x) = 2*sigmoid(2x) - 1, same saturation argument applies.
__device__ __forceinline__ float fast_tanh(float x) {
  return 2.0f * __builtin_amdgcn_rcpf(
                    1.0f + __builtin_amdgcn_exp2f(-2.885390081777927f * x)) -
         1.0f;
}

// mlp: p = tanh(h @ w1^T + b1) @ w2^T + b2 ; writes pred_x and fp16 p buffer.
__device__ __forceinline__ void mlp_step(const _Float16* hT, _Float16* aT,
                                         _Float16* pT, const _Float16* sW1,
                                         const _Float16* sW2, const float* sB1,
                                         const float* sB2,
                                         float* __restrict__ predx, int row0,
                                         int t, int lane) {
  const v8f vzero = {0.f, 0.f, 0.f, 0.f, 0.f, 0.f, 0.f, 0.f};
  const int laneN = lane & 15;
  const int mh = (lane >> 4) * 8;

  v16h hA[4];
#pragma unroll
  for (int kt = 0; kt < 4; ++kt) hA[kt] = load_A(hT, LDIM, kt * 32, lane);

  // stage 1: a1 = tanh(h @ w1^T + b1), 8 N-tiles, K=128
  for (int j = 0; j < 8; ++j) {
    v8f acc = vzero;
#pragma unroll
    for (int kt = 0; kt < 4; ++kt)
      acc = wmma16(hA[kt], load_B(sW1, LDIM, j * 16, kt * 32, lane), acc);
    const int col = j * 16 + laneN;
    const float bb = sB1[col];
#pragma unroll
    for (int v = 0; v < 8; ++v)
      aT[(v + mh) * UDIM + col] = (_Float16)fast_tanh(acc[v] + bb);
  }
  __syncthreads();

  // stage 2: p = a1 @ w2^T + b2, 4 N-tiles, K=128
  v16h aA[4];
#pragma unroll
  for (int kt = 0; kt < 4; ++kt) aA[kt] = load_A(aT, UDIM, kt * 32, lane);
  for (int j = 0; j < 4; ++j) {
    v8f acc = vzero;
#pragma unroll
    for (int kt = 0; kt < 4; ++kt)
      acc = wmma16(aA[kt], load_B(sW2, UDIM, j * 16, kt * 32, lane), acc);
    const int col = j * 16 + laneN;
    const float bb = sB2[col];
#pragma unroll
    for (int v = 0; v < 8; ++v) {
      const int m = v + mh;
      const float pv = acc[v] + bb;
      predx[((size_t)(row0 + m) * TSTEPS + t) * DDIM + col] = pv;
      pT[m * DDIM + col] = (_Float16)pv;
    }
  }
}

extern "C" __global__ void __launch_bounds__(WAVES * 32, 1)
rnn_decoder(const float* __restrict__ first_point, const float* __restrict__ ts,
            const float* __restrict__ w_ih, const float* __restrict__ w_hh,
            const float* __restrict__ b_ih, const float* __restrict__ b_hh,
            const float* __restrict__ w1, const float* __restrict__ b1,
            const float* __restrict__ w2, const float* __restrict__ b2,
            float* __restrict__ out) {
  // Weights staged once as fp16 (ISA: up to 320KB LDS per workgroup).
  __shared__ _Float16 sWih[GDIM * DDIM];   // 48 KB  (first 64 cols of w_ih)
  __shared__ _Float16 sWhh[GDIM * LDIM];   // 96 KB
  __shared__ _Float16 sW1[UDIM * LDIM];    // 32 KB
  __shared__ _Float16 sW2[DDIM * UDIM];    // 16 KB
  __shared__ float sWdt[GDIM];             // dt column (col 64) of w_ih
  __shared__ float sBih[GDIM], sBhh[GDIM], sB1[UDIM], sB2[DDIM];
  __shared__ _Float16 hbuf[WAVES][2][WROWS * LDIM];  // 32 KB ping-pong h (+a1)
  __shared__ _Float16 pbuf[WAVES][WROWS * DDIM];     // 8 KB  fp16 p

  const int tid = threadIdx.x;
  const int lane = tid & 31;
  const int wave = tid >> 5;
  const int row0 = blockIdx.x * WGROWS + wave * WROWS;
  float* predx = out + (size_t)BATCH * TSTEPS * LDIM;
  const v8f vzero = {0.f, 0.f, 0.f, 0.f, 0.f, 0.f, 0.f, 0.f};

  // ---- cooperative weight staging (f32 -> f16) ----
  for (int i = tid; i < GDIM * DDIM; i += WAVES * 32)
    sWih[i] = (_Float16)w_ih[(i >> 6) * (DDIM + 1) + (i & 63)];
  for (int i = tid; i < GDIM; i += WAVES * 32) {
    sWdt[i] = w_ih[i * (DDIM + 1) + DDIM];
    sBih[i] = b_ih[i];
    sBhh[i] = b_hh[i];
  }
  for (int i = tid; i < GDIM * LDIM; i += WAVES * 32) sWhh[i] = (_Float16)w_hh[i];
  for (int i = tid; i < UDIM * LDIM; i += WAVES * 32) sW1[i] = (_Float16)w1[i];
  for (int i = tid; i < DDIM * UDIM; i += WAVES * 32) sW2[i] = (_Float16)w2[i];
  for (int i = tid; i < UDIM; i += WAVES * 32) sB1[i] = b1[i];
  for (int i = tid; i < DDIM; i += WAVES * 32) sB2[i] = b2[i];

  // ---- h0: write sol_z[t=0] and stage fp16 h into hbuf[1] ----
  _Float16* pb = pbuf[wave];
  for (int i = lane; i < WROWS * LDIM; i += 32) {
    const int m = i >> 7, c = i & 127;
    const float v = first_point[(size_t)(row0 + m) * LDIM + c];
    out[((size_t)(row0 + m) * TSTEPS + 0) * LDIM + c] = v;
    hbuf[wave][1][m * LDIM + c] = (_Float16)v;
  }
  __syncthreads();

  // p0 = mlp(h0)  (uses hbuf[0] as tanh scratch)
  mlp_step(hbuf[wave][1], hbuf[wave][0], pb, sW1, sW2, sB1, sB2, predx, row0, 0,
           lane);

  const int laneN = lane & 15;
  const int mh = (lane >> 4) * 8;
  int cur = 1;
  for (int t = 1; t < TSTEPS; ++t) {
    __syncthreads();  // p (pbuf) from previous mlp is visible
    const float dt = ts[t] - ts[t - 1];
    _Float16* hOld = hbuf[wave][cur];
    _Float16* hNew = hbuf[wave][cur ^ 1];

    // Hoist A operands for the whole step: p is 16x64, h is 16x128.
    v16h pA[2], hA[4];
#pragma unroll
    for (int kt = 0; kt < 2; ++kt) pA[kt] = load_A(pb, DDIM, kt * 32, lane);
#pragma unroll
    for (int kt = 0; kt < 4; ++kt) hA[kt] = load_A(hOld, LDIM, kt * 32, lane);

    // 8 column tiles of the hidden dim; r/z fused, n kept as i_n + r*h_n.
    for (int j = 0; j < 8; ++j) {
      v8f ar = vzero, az = vzero, ani = vzero, anh = vzero;
#pragma unroll
      for (int kt = 0; kt < 2; ++kt) {  // gi = p @ w_ih[:, :64]^T  (K=64)
        const v16h br_ = load_B(sWih, DDIM, j * 16, kt * 32, lane);
        const v16h bz_ = load_B(sWih, DDIM, LDIM + j * 16, kt * 32, lane);
        const v16h bn_ = load_B(sWih, DDIM, 2 * LDIM + j * 16, kt * 32, lane);
        ar = wmma16(pA[kt], br_, ar);
        az = wmma16(pA[kt], bz_, az);
        ani = wmma16(pA[kt], bn_, ani);
      }
#pragma unroll
      for (int kt = 0; kt < 4; ++kt) {  // gh = h @ w_hh^T  (K=128)
        const v16h br_ = load_B(sWhh, LDIM, j * 16, kt * 32, lane);
        const v16h bz_ = load_B(sWhh, LDIM, LDIM + j * 16, kt * 32, lane);
        const v16h bn_ = load_B(sWhh, LDIM, 2 * LDIM + j * 16, kt * 32, lane);
        ar = wmma16(hA[kt], br_, ar);
        az = wmma16(hA[kt], bz_, az);
        anh = wmma16(hA[kt], bn_, anh);
      }
      // gates + state update; dt column folded into the bias terms.
      const int col = j * 16 + laneN;
      const float br = sBih[col] + sBhh[col] + dt * sWdt[col];
      const float bz =
          sBih[LDIM + col] + sBhh[LDIM + col] + dt * sWdt[LDIM + col];
      const float bin = sBih[2 * LDIM + col] + dt * sWdt[2 * LDIM + col];
      const float bhn = sBhh[2 * LDIM + col];
#pragma unroll
      for (int v = 0; v < 8; ++v) {
        const int m = v + mh;
        const float r = fast_sigmoid(ar[v] + br);
        const float z = fast_sigmoid(az[v] + bz);
        const float n = fast_tanh(ani[v] + bin + r * (anh[v] + bhn));
        const float hold = (float)hOld[m * LDIM + col];
        const float hnew = (1.0f - z) * n + z * hold;
        out[((size_t)(row0 + m) * TSTEPS + t) * LDIM + col] = hnew;
        hNew[m * LDIM + col] = (_Float16)hnew;
      }
    }
    __syncthreads();  // hNew complete; hOld free -> reuse as tanh scratch
    mlp_step(hNew, hOld, pb, sW1, sW2, sB1, sB2, predx, row0, t, lane);
    cur ^= 1;
  }
}

extern "C" void kernel_launch(void* const* d_in, const int* in_sizes, int n_in,
                              void* d_out, int out_size, void* d_ws,
                              size_t ws_size, hipStream_t stream) {
  const float* first_point = (const float*)d_in[0];
  const float* ts          = (const float*)d_in[1];
  const float* w_ih        = (const float*)d_in[2];
  const float* w_hh        = (const float*)d_in[3];
  const float* b_ih        = (const float*)d_in[4];
  const float* b_hh        = (const float*)d_in[5];
  const float* w1          = (const float*)d_in[6];
  const float* b1          = (const float*)d_in[7];
  const float* w2          = (const float*)d_in[8];
  const float* b2          = (const float*)d_in[9];
  float* out = (float*)d_out;

  dim3 grid(BATCH / WGROWS);   // 32 workgroups
  dim3 block(WAVES * 32);      // 4 waves (wave32)
  rnn_decoder<<<grid, block, 0, stream>>>(first_point, ts, w_ih, w_hh, b_ih,
                                          b_hh, w1, b1, w2, b2, out);
}
